// ConvMHSA_76081050682039
// MI455X (gfx1250) — matrix operand order repaired and checked by reading, not compiled, not run on hardware
//
#include <hip/hip_runtime.h>

// ---------------------------------------------------------------------------
// ConvMHSA for MI455X (gfx1250), wave32 + WMMA bf16 16x16x32, f32 accumulate.
//  x:(16,512,32,32) f32 -> out same shape.
//  Pipeline: [cvt/transpose] -> [QKV GEMM] -> [flash attention] -> [out GEMM]
// ---------------------------------------------------------------------------

typedef unsigned short u16;
typedef __attribute__((ext_vector_type(16))) __bf16 v16bf;
typedef __attribute__((ext_vector_type(8)))  float  v8f;
typedef uint4 __attribute__((may_alias)) uint4_ma;

union FragAB {        // 16 bf16 per lane = one WMMA A or B operand
    v16bf    v;
    uint4_ma q[2];
};

__device__ __forceinline__ u16 f32_to_bf16(float f) {
    unsigned u = __float_as_uint(f);
    u += 0x7FFFu + ((u >> 16) & 1u);          // round-to-nearest-even
    return (u16)(u >> 16);
}

__device__ __forceinline__ v8f wmma_bf16(const FragAB& a, const FragAB& b, v8f c) {
    return __builtin_amdgcn_wmma_f32_16x16x32_bf16(
        false, a.v, false, b.v, (short)0, c, false, false);
}

#define LD16(dst, p)  (dst) = *(const uint4_ma*)(p)

// ---------------------------------------------------------------------------
// Prep: f32 -> bf16 straight convert (weights)
// ---------------------------------------------------------------------------
__global__ void cvt_bf16_kernel(const float* __restrict__ src, u16* __restrict__ dst, int n) {
    int i = blockIdx.x * blockDim.x + threadIdx.x;
    if (i < n) dst[i] = f32_to_bf16(src[i]);
}

// Prep: x (B,C,N) f32 -> xt (B,N,C) bf16, 32x32 LDS-tiled transpose
__global__ void xpose_cvt_kernel(const float* __restrict__ x, u16* __restrict__ xt) {
    __shared__ float tile[32][33];
    const int b  = blockIdx.z;
    const int c0 = blockIdx.y * 32;
    const int n0 = blockIdx.x * 32;
    const float* xb = x + (size_t)b * 512 * 1024;
    for (int i = threadIdx.y; i < 32; i += 8)
        tile[i][threadIdx.x] = xb[(size_t)(c0 + i) * 1024 + n0 + threadIdx.x];
    __syncthreads();
    u16* xtb = xt + (size_t)b * 1024 * 512;
    for (int i = threadIdx.y; i < 32; i += 8)
        xtb[(size_t)(n0 + i) * 512 + c0 + threadIdx.x] = f32_to_bf16(tile[threadIdx.x][i]);
}

// ---------------------------------------------------------------------------
// QKV GEMM:  qkv[o,n] = sum_c w[o,c] * x[c,n] + bias[o]   (per batch)
//  A = qkv_w bf16 (1536x512 row-major), B = xt (B,N,C) bf16.
//  Block = 4 waves, each wave a 64x64 tile (2x2 wave grid -> 128x128/block).
//  Scatter: q,k -> (BH, N, 64) bf16 ; v -> (BH, 64, N) bf16.
// ---------------------------------------------------------------------------
__global__ __launch_bounds__(128) void qkv_gemm_kernel(
    const u16* __restrict__ w, const float* __restrict__ bias,
    const u16* __restrict__ xt,
    u16* __restrict__ qT, u16* __restrict__ kT, u16* __restrict__ vv)
{
    const int wave = threadIdx.x >> 5, lane = threadIdx.x & 31;
    const int half = lane >> 4, l16 = lane & 15;
    const int b    = blockIdx.z;
    const int o64  = blockIdx.y * 128 + (wave >> 1) * 64;
    const int n64  = blockIdx.x * 128 + (wave & 1) * 64;
    const u16* xb  = xt + (size_t)b * 1024 * 512;

    v8f acc[4][4];
#pragma unroll
    for (int i = 0; i < 4; ++i)
#pragma unroll
        for (int j = 0; j < 4; ++j) acc[i][j] = (v8f)0.0f;

    for (int kk = 0; kk < 512; kk += 32) {
        FragAB a[4], bf[4];
#pragma unroll
        for (int i = 0; i < 4; ++i) {
            const u16* ap = w + (size_t)(o64 + i * 16 + l16) * 512 + kk;
            LD16(a[i].q[0], ap + 8 * half);
            LD16(a[i].q[1], ap + 16 + 8 * half);
        }
#pragma unroll
        for (int j = 0; j < 4; ++j) {
            const u16* bp = xb + (size_t)(n64 + j * 16 + l16) * 512 + kk;
            LD16(bf[j].q[0], bp + 8 * half);
            LD16(bf[j].q[1], bp + 16 + 8 * half);
        }
        __builtin_prefetch(w + (size_t)(o64 + l16) * 512 + kk + 32, 0, 3);
        __builtin_prefetch(xb + (size_t)(n64 + l16) * 512 + kk + 32, 0, 3);
#pragma unroll
        for (int i = 0; i < 4; ++i)
#pragma unroll
            for (int j = 0; j < 4; ++j)
                acc[i][j] = wmma_bf16(a[i], bf[j], acc[i][j]);
    }

    // epilogue: + bias, convert to bf16, scatter into q/k/v attention layouts
#pragma unroll
    for (int i = 0; i < 4; ++i) {
#pragma unroll
        for (int r = 0; r < 8; ++r) {
            const int o  = o64 + i * 16 + r + 8 * half;
            const float bvv = bias[o];
#pragma unroll
            for (int j = 0; j < 4; ++j) {
                const int n = n64 + j * 16 + l16;
                const u16 h = f32_to_bf16(acc[i][j][r] + bvv);
                if (o < 512) {
                    const int hb = b * 8 + (o >> 6);
                    qT[((size_t)hb * 1024 + n) * 64 + (o & 63)] = h;
                } else if (o < 1024) {
                    const int o2 = o - 512, hb = b * 8 + (o2 >> 6);
                    kT[((size_t)hb * 1024 + n) * 64 + (o2 & 63)] = h;
                } else {
                    const int o2 = o - 1024, hb = b * 8 + (o2 >> 6);
                    vv[((size_t)hb * 64 + (o2 & 63)) * 1024 + n] = h;
                }
            }
        }
    }
}

// ---------------------------------------------------------------------------
// Flash attention: per wave, 16 queries x all 1024 keys, online softmax over
// 32-key chunks; P goes through a 1KB per-wave LDS tile (D-layout -> A-frag).
// Output -> attnT (B, N, 512) bf16 for the final projection GEMM.
// ---------------------------------------------------------------------------
__global__ __launch_bounds__(128) void attn_kernel(
    const u16* __restrict__ qT, const u16* __restrict__ kT,
    const u16* __restrict__ vv, u16* __restrict__ attnT)
{
    __shared__ __align__(16) u16 sP[4][16 * 32];   // per-wave 16x32 P tile
    const int wave = threadIdx.x >> 5, lane = threadIdx.x & 31;
    const int half = lane >> 4, l16 = lane & 15;
    const int hb = blockIdx.y;                      // head-batch 0..127
    const int q0 = blockIdx.x * 64 + wave * 16;     // this wave's 16 queries

    const u16* qb = qT + (size_t)hb * 1024 * 64;
    const u16* kb = kT + (size_t)hb * 1024 * 64;
    const u16* vb = vv + (size_t)hb * 64 * 1024;

    // Q A-fragments for the two K-steps of hd=64; loaded once, reused 64x.
    FragAB aq0, aq1;
    {
        const u16* qp = qb + (size_t)(q0 + l16) * 64;
        LD16(aq0.q[0], qp + 8 * half);
        LD16(aq0.q[1], qp + 16 + 8 * half);
        LD16(aq1.q[0], qp + 32 + 8 * half);
        LD16(aq1.q[1], qp + 48 + 8 * half);
    }

    v8f o[4];
#pragma unroll
    for (int t = 0; t < 4; ++t) o[t] = (v8f)0.0f;
    float rmax[8], rsum[8];
#pragma unroll
    for (int r = 0; r < 8; ++r) { rmax[r] = -1e30f; rsum[r] = 0.0f; }

    const float sc = 0.125f;                        // 1/sqrt(64)
    for (int k0 = 0; k0 < 1024; k0 += 32) {
        // ---- scores for key tiles [k0,k0+16) and [k0+16,k0+32)
        v8f s0 = (v8f)0.0f, s1 = (v8f)0.0f;
#pragma unroll
        for (int t = 0; t < 2; ++t) {
            FragAB b0, b1;
            const u16* kp = kb + (size_t)(k0 + t * 16 + l16) * 64;
            LD16(b0.q[0], kp + 8 * half);
            LD16(b0.q[1], kp + 16 + 8 * half);
            LD16(b1.q[0], kp + 32 + 8 * half);
            LD16(b1.q[1], kp + 48 + 8 * half);
            if (t == 0) { s0 = wmma_bf16(aq0, b0, s0); s0 = wmma_bf16(aq1, b1, s0); }
            else        { s1 = wmma_bf16(aq0, b0, s1); s1 = wmma_bf16(aq1, b1, s1); }
        }

        // ---- online softmax, rows m = r + 8*half live in this 16-lane group
#pragma unroll
        for (int r = 0; r < 8; ++r) {
            float v0 = s0[r] * sc, v1 = s1[r] * sc;
            float m = fmaxf(v0, v1);
#pragma unroll
            for (int msk = 1; msk < 16; msk <<= 1)
                m = fmaxf(m, __shfl_xor(m, msk, 32));
            const float mn = fmaxf(rmax[r], m);
            const float scale = __expf(rmax[r] - mn);
            const float p0 = __expf(v0 - mn);
            const float p1 = __expf(v1 - mn);
            float ps = p0 + p1;
#pragma unroll
            for (int msk = 1; msk < 16; msk <<= 1)
                ps += __shfl_xor(ps, msk, 32);
            rsum[r] = rsum[r] * scale + ps;
            rmax[r] = mn;
#pragma unroll
            for (int t = 0; t < 4; ++t) o[t][r] *= scale;
            const int m16 = r + 8 * half;
            sP[wave][m16 * 32 + l16]      = f32_to_bf16(p0);
            sP[wave][m16 * 32 + 16 + l16] = f32_to_bf16(p1);
        }
        asm volatile("" ::: "memory");   // keep DS store -> DS load in order

        // ---- P tile back as A-fragment (row-major [m][k] in LDS)
        FragAB ap;
        {
            const u16* pp = &sP[wave][l16 * 32];
            LD16(ap.q[0], pp + 8 * half);
            LD16(ap.q[1], pp + 16 + 8 * half);
        }

        // ---- O += P x V^T : B[k=key][n=c] from v (hd,N) row-major in keys
#pragma unroll
        for (int t = 0; t < 4; ++t) {
            FragAB bv;
            const u16* vp = vb + (size_t)(t * 16 + l16) * 1024 + k0;
            LD16(bv.q[0], vp + 8 * half);
            LD16(bv.q[1], vp + 16 + 8 * half);
            o[t] = wmma_bf16(ap, bv, o[t]);
        }
    }

    // ---- normalize & store to attnT (B, N, 512)
    const int b = hb >> 3, head = hb & 7;
#pragma unroll
    for (int t = 0; t < 4; ++t)
#pragma unroll
        for (int r = 0; r < 8; ++r) {
            const int m = r + 8 * half;
            const int n = q0 + m;
            const int c = head * 64 + t * 16 + l16;
            attnT[((size_t)b * 1024 + n) * 512 + c] = f32_to_bf16(o[t][r] / rsum[r]);
        }
}

// ---------------------------------------------------------------------------
// Output projection + fused residual:
//   out[o,n] = gamma * (sum_c W[o,c]*attn[c,n] + out_b[o]) + x[o,n]   (f32)
// ---------------------------------------------------------------------------
__global__ __launch_bounds__(128) void out_gemm_kernel(
    const u16* __restrict__ w, const float* __restrict__ bias,
    const float* __restrict__ gamma, const u16* __restrict__ attnT,
    const float* __restrict__ x, float* __restrict__ out)
{
    const int wave = threadIdx.x >> 5, lane = threadIdx.x & 31;
    const int half = lane >> 4, l16 = lane & 15;
    const int b    = blockIdx.z;
    const int o64  = blockIdx.y * 128 + (wave >> 1) * 64;
    const int n64  = blockIdx.x * 128 + (wave & 1) * 64;
    const u16* ab  = attnT + (size_t)b * 1024 * 512;

    v8f acc[4][4];
#pragma unroll
    for (int i = 0; i < 4; ++i)
#pragma unroll
        for (int j = 0; j < 4; ++j) acc[i][j] = (v8f)0.0f;

    for (int kk = 0; kk < 512; kk += 32) {
        FragAB a[4], bf[4];
#pragma unroll
        for (int i = 0; i < 4; ++i) {
            const u16* ap = w + (size_t)(o64 + i * 16 + l16) * 512 + kk;
            LD16(a[i].q[0], ap + 8 * half);
            LD16(a[i].q[1], ap + 16 + 8 * half);
        }
#pragma unroll
        for (int j = 0; j < 4; ++j) {
            const u16* bp = ab + (size_t)(n64 + j * 16 + l16) * 512 + kk;
            LD16(bf[j].q[0], bp + 8 * half);
            LD16(bf[j].q[1], bp + 16 + 8 * half);
        }
        __builtin_prefetch(w + (size_t)(o64 + l16) * 512 + kk + 32, 0, 3);
        __builtin_prefetch(ab + (size_t)(n64 + l16) * 512 + kk + 32, 0, 3);
#pragma unroll
        for (int i = 0; i < 4; ++i)
#pragma unroll
            for (int j = 0; j < 4; ++j)
                acc[i][j] = wmma_bf16(a[i], bf[j], acc[i][j]);
    }

    const float g = gamma[0];
#pragma unroll
    for (int i = 0; i < 4; ++i)
#pragma unroll
        for (int r = 0; r < 8; ++r) {
            const int o = o64 + i * 16 + r + 8 * half;
            const float bvv = bias[o];
#pragma unroll
            for (int j = 0; j < 4; ++j) {
                const int n = n64 + j * 16 + l16;
                const size_t idx = ((size_t)b * 512 + o) * 1024 + n;
                out[idx] = g * (acc[i][j][r] + bvv) + x[idx];
            }
        }
}

// ---------------------------------------------------------------------------
extern "C" void kernel_launch(void* const* d_in, const int* in_sizes, int n_in,
                              void* d_out, int out_size, void* d_ws, size_t ws_size,
                              hipStream_t stream) {
    const float* x      = (const float*)d_in[0];
    const float* qkv_w  = (const float*)d_in[1];
    const float* qkv_b  = (const float*)d_in[2];
    const float* out_w  = (const float*)d_in[3];
    const float* out_b  = (const float*)d_in[4];
    const float* gamma  = (const float*)d_in[5];
    float*       out    = (float*)d_out;

    char* ws = (char*)d_ws;
    size_t off = 0;
    u16* xt   = (u16*)(ws + off); off += (size_t)16 * 1024 * 512 * 2;  // (B,N,C)
    u16* wqb  = (u16*)(ws + off); off += (size_t)1536 * 512 * 2;       // qkv_w bf16
    u16* wob  = (u16*)(ws + off); off += (size_t)512 * 512 * 2;        // out_w bf16
    u16* qT   = (u16*)(ws + off); off += (size_t)128 * 1024 * 64 * 2;  // (BH,N,hd)
    u16* kT   = (u16*)(ws + off); off += (size_t)128 * 1024 * 64 * 2;  // (BH,N,hd)
    u16* vv   = (u16*)(ws + off); off += (size_t)128 * 64 * 1024 * 2;  // (BH,hd,N)
    u16* aT   = (u16*)(ws + off); off += (size_t)16 * 1024 * 512 * 2;  // (B,N,C)

    // Stage 0: precision conversion / layout
    xpose_cvt_kernel<<<dim3(32, 16, 16), dim3(32, 8), 0, stream>>>(x, xt);
    cvt_bf16_kernel<<<(1536 * 512 + 255) / 256, 256, 0, stream>>>(qkv_w, wqb, 1536 * 512);
    cvt_bf16_kernel<<<(512 * 512 + 255) / 256, 256, 0, stream>>>(out_w, wob, 512 * 512);

    // Stage 1: QKV projection
    qkv_gemm_kernel<<<dim3(8, 12, 16), 128, 0, stream>>>(wqb, qkv_b, xt, qT, kT, vv);

    // Stage 2: attention (flash, online softmax)
    attn_kernel<<<dim3(16, 128), 128, 0, stream>>>(qT, kT, vv, aT);

    // Stage 3: output projection + gamma*(.)+residual
    out_gemm_kernel<<<dim3(8, 4, 16), 128, 0, stream>>>(wob, out_b, gamma, aT, x, out);
}
